// VisionMambaDenoiser_29205777613724
// MI455X (gfx1250) — compile-verified, matrix operand order checked by reading, not run on hardware
//
#include <hip/hip_runtime.h>
#include <hip/hip_bf16.h>
#include <math.h>

typedef __attribute__((ext_vector_type(16))) _Float16 v16h;
typedef __attribute__((ext_vector_type(8)))  float    v8f;
typedef unsigned int u32x4 __attribute__((ext_vector_type(4)));
typedef int          i32x8 __attribute__((ext_vector_type(8)));
typedef int          i32x4 __attribute__((ext_vector_type(4)));

#define D_MODEL 192
#define D_INNER 384
#define D_STATE 16
#define DT_RANK 12
#define SEQLEN  1024

#if defined(__has_builtin)
#  if __has_builtin(__builtin_amdgcn_tensor_load_to_lds) && \
      __has_builtin(__builtin_amdgcn_s_wait_tensorcnt)
#    define HAVE_TDM 1
#  endif
#endif
#ifndef HAVE_TDM
#  define HAVE_TDM 0
#endif

enum { A_PLAIN = 0, A_IM2COL = 1 };
enum { EPI_STORE = 0, EPI_STORE_BIAS = 1, EPI_DECONV = 2 };
enum { B_NK = 0, B_KN = 1 };

// ---- staging helpers: global->regs (issue early) and regs->LDS (f32->f16) --
template <int AMODE>
__device__ __forceinline__ void gemm_load_a(
    const float* __restrict__ A, int m0, int k0, int K, int tid, float4 (&va)[4])
{
  const int arow = tid >> 3;           // 0..31
  const int acol = (tid & 7) << 2;     // 0..28 step 4
  #pragma unroll
  for (int rr = 0; rr < 4; ++rr) {
    const int gm = m0 + rr * 32 + arow;
    const int gk = k0 + acol;
    if (AMODE == A_PLAIN) {
      va[rr] = *reinterpret_cast<const float4*>(A + (size_t)gm * K + gk);
    } else {
      const int b  = gm >> 10;
      const int hw = gm & 1023;
      const int ph = hw >> 5, pw = hw & 31;
      const int c  = gk >> 8;
      const int rm = gk & 255;
      const int py = rm >> 4, px = rm & 15;
      va[rr] = *reinterpret_cast<const float4*>(
          A + (((size_t)(b * 3 + c) * 512 + ph * 16 + py) * 512 + pw * 16 + px));
    }
  }
}

__device__ __forceinline__ void gemm_store_a(
    _Float16 (*As)[40], int tid, const float4 (&va)[4])
{
  const int arow = tid >> 3;
  const int acol = (tid & 7) << 2;
  #pragma unroll
  for (int rr = 0; rr < 4; ++rr) {
    _Float16 h4[4];
    h4[0] = (_Float16)va[rr].x; h4[1] = (_Float16)va[rr].y;
    h4[2] = (_Float16)va[rr].z; h4[3] = (_Float16)va[rr].w;
    *reinterpret_cast<uint2*>(&As[rr * 32 + arow][acol]) =
        *reinterpret_cast<const uint2*>(h4);
  }
}

template <int BLAYOUT>
__device__ __forceinline__ void gemm_load_b(
    const float* __restrict__ Bw, int n0, int k0, int N, int ldb, int tid,
    float4 (&vb)[2])
{
  if (BLAYOUT == B_NK) {
    const int bn = tid >> 2;           // 0..63
    const int bk = (tid & 3) << 3;     // 0,8,16,24
    const int gn = n0 + bn;
    if (gn < N) {
      const float* p = Bw + (size_t)gn * ldb + (k0 + bk);
      vb[0] = *reinterpret_cast<const float4*>(p);
      vb[1] = *reinterpret_cast<const float4*>(p + 4);
    } else {
      vb[0] = make_float4(0.f, 0.f, 0.f, 0.f);
      vb[1] = make_float4(0.f, 0.f, 0.f, 0.f);
    }
  } else {
    const int bk = tid >> 3;           // 0..31
    const int nq = (tid & 7) << 3;     // 0..56
    const float* p = Bw + (size_t)(k0 + bk) * ldb + (n0 + nq);
    vb[0] = *reinterpret_cast<const float4*>(p);
    vb[1] = *reinterpret_cast<const float4*>(p + 4);
  }
}

template <int BLAYOUT>
__device__ __forceinline__ void gemm_store_b(
    _Float16 (*Bs)[40], int tid, const float4 (&vb)[2])
{
  if (BLAYOUT == B_NK) {
    const int bn = tid >> 2;
    const int bk = (tid & 3) << 3;
    _Float16 h8[8];
    h8[0] = (_Float16)vb[0].x; h8[1] = (_Float16)vb[0].y;
    h8[2] = (_Float16)vb[0].z; h8[3] = (_Float16)vb[0].w;
    h8[4] = (_Float16)vb[1].x; h8[5] = (_Float16)vb[1].y;
    h8[6] = (_Float16)vb[1].z; h8[7] = (_Float16)vb[1].w;
    *reinterpret_cast<uint4*>(&Bs[bn][bk]) =
        *reinterpret_cast<const uint4*>(h8);
  } else {
    const int bk = tid >> 3;
    const int nq = (tid & 7) << 3;
    Bs[nq + 0][bk] = (_Float16)vb[0].x; Bs[nq + 1][bk] = (_Float16)vb[0].y;
    Bs[nq + 2][bk] = (_Float16)vb[0].z; Bs[nq + 3][bk] = (_Float16)vb[0].w;
    Bs[nq + 4][bk] = (_Float16)vb[1].x; Bs[nq + 5][bk] = (_Float16)vb[1].y;
    Bs[nq + 6][bk] = (_Float16)vb[1].z; Bs[nq + 7][bk] = (_Float16)vb[1].w;
  }
}

// ---------------------------------------------------------------------------
// Generic WMMA GEMM: C[M x N] = A[M x K] * B[K x N]
//   Double-buffered LDS pipeline: global loads for tile k+1 issue before the
//   WMMAs of tile k; one barrier per iteration. BM=128 BN=64 BK=32, 8 waves,
//   wave tile 32x32 = 2x2 v_wmma_f32_16x16x32_f16 (f32 accumulate).
// ---------------------------------------------------------------------------
template <int AMODE, int EPI, int BLAYOUT>
__global__ __launch_bounds__(256) void gemm_wmma(
    const float* __restrict__ A, const float* __restrict__ Bw,
    float* __restrict__ C, const float* __restrict__ bias,
    int M, int N, int K, int ldc, int ldb)
{
  __shared__ _Float16 As[2][128][40];   // row stride 80B (16B-aligned runs)
  __shared__ _Float16 Bs[2][64][40];    // TRANSPOSED: [n][k]
  const int tid   = threadIdx.x;
  const int lane  = tid & 31;
  const int wave  = tid >> 5;
  const int wm    = wave >> 1;      // 0..3
  const int wn    = wave & 1;       // 0..1
  const int lhalf = lane >> 4;
  const int lmod  = lane & 15;
  const int m0    = blockIdx.y * 128;
  const int n0    = blockIdx.x * 64;

  v8f acc[2][2] = {};
  const int ktiles = K >> 5;

  float4 va[4], vb[2];
  gemm_load_a<AMODE>(A, m0, 0, K, tid, va);
  gemm_load_b<BLAYOUT>(Bw, n0, 0, N, ldb, tid, vb);
  gemm_store_a(As[0], tid, va);
  gemm_store_b<BLAYOUT>(Bs[0], tid, vb);
  __syncthreads();

  for (int kt = 0; kt < ktiles; ++kt) {
    const int cur = kt & 1;
    const bool more = (kt + 1 < ktiles);
    if (more) {
      const int k1 = (kt + 1) << 5;
      gemm_load_a<AMODE>(A, m0, k1, K, tid, va);       // in flight during WMMA
      gemm_load_b<BLAYOUT>(Bw, n0, k1, N, ldb, tid, vb);
      if (kt + 2 < ktiles) {  // global_prefetch_b8 hints two tiles ahead
        if (AMODE == A_PLAIN)
          __builtin_prefetch(A + (size_t)(m0 + (tid & 127)) * K + k1 + 32, 0, 1);
        if (BLAYOUT == B_NK)
          __builtin_prefetch(Bw + (size_t)(n0 + (tid & 63)) * ldb + k1 + 32, 0, 1);
        else
          __builtin_prefetch(Bw + (size_t)(k1 + 32 + (tid & 31)) * ldb + n0, 0, 1);
      }
    }

    // ---- fragments (16-bit A 16x32 / B 32x16 VGPR striping) ----
    v16h afrag[2], bfrag[2];
    #pragma unroll
    for (int mt = 0; mt < 2; ++mt) {
      const int row = wm * 32 + mt * 16 + lmod;
      #pragma unroll
      for (int e = 0; e < 16; ++e) {
        const int kk = ((e < 8) ? 0 : 16) + lhalf * 8 + (e & 7);
        afrag[mt][e] = As[cur][row][kk];
      }
    }
    #pragma unroll
    for (int nt = 0; nt < 2; ++nt) {
      const int col = wn * 32 + nt * 16 + lmod;
      #pragma unroll
      for (int e = 0; e < 16; ++e) {
        const int kk = ((e < 8) ? 0 : 16) + lhalf * 8 + (e & 7);
        bfrag[nt][e] = Bs[cur][col][kk];
      }
    }
    #pragma unroll
    for (int mt = 0; mt < 2; ++mt)
      #pragma unroll
      for (int nt = 0; nt < 2; ++nt)
        acc[mt][nt] = __builtin_amdgcn_wmma_f32_16x16x32_f16(
            false, afrag[mt], false, bfrag[nt], (short)0, acc[mt][nt],
            false, false);

    if (more) {
      gemm_store_a(As[cur ^ 1], tid, va);
      gemm_store_b<BLAYOUT>(Bs[cur ^ 1], tid, vb);
    }
    __syncthreads();
  }

  // ---- epilogue (C/D layout: lane%16 = N, M = r + 8*(lane/16)) ----
  #pragma unroll
  for (int mt = 0; mt < 2; ++mt) {
    #pragma unroll
    for (int nt = 0; nt < 2; ++nt) {
      const int gmb = m0 + wm * 32 + mt * 16 + lhalf * 8;
      const int gn  = n0 + wn * 32 + nt * 16 + lmod;
      #pragma unroll
      for (int r = 0; r < 8; ++r) {
        const float v  = acc[mt][nt][r];
        const int   gm = gmb + r;
        if (EPI == EPI_DECONV) {
          const int h = gm >> 5, w = gm & 31;
          const int dch = gn >> 8, pq = gn & 255;
          const int p = pq >> 4, q = pq & 15;
          float o = v + bias[dch];
          o = o > 0.f ? o : 0.f;  // fused ReLU
          C[((size_t)dch * 512 + h * 16 + p) * 512 + (w * 16 + q)] = o;
        } else {
          if (gn < N) {
            float o = v;
            if (EPI == EPI_STORE_BIAS) o += bias[gn];
            C[(size_t)gm * ldc + gn] = o;
          }
        }
      }
    }
  }
}

// ---------------------------------------------------------------------------
// depthwise causal conv1d (k=4) + bias + SiLU
// ---------------------------------------------------------------------------
__global__ __launch_bounds__(256) void conv1d_silu_k(
    const float* __restrict__ xz, const float* __restrict__ w,
    const float* __restrict__ bia, float* __restrict__ uact)
{
  const int i  = blockIdx.x * 256 + threadIdx.x;
  const int d  = i % D_INNER;
  const int bl = i / D_INNER;
  const int b  = bl >> 10, l = bl & 1023;
  float acc = bia[d];
  #pragma unroll
  for (int j = 0; j < 4; ++j) {
    const int ls = l - 3 + j;
    if (ls >= 0)
      acc += xz[((size_t)(b * SEQLEN + ls)) * (2 * D_INNER) + d] * w[d * 4 + j];
  }
  uact[i] = acc / (1.f + __expf(-acc));
}

// ---------------------------------------------------------------------------
// delta = softplus(dt @ dt_proj_w^T + dt_proj_b)  (K=12, VALU)
// ---------------------------------------------------------------------------
__global__ __launch_bounds__(256) void delta_k(
    const float* __restrict__ dbc, const float* __restrict__ dtw,
    const float* __restrict__ dtb, float* __restrict__ dlt)
{
  const int i  = blockIdx.x * 256 + threadIdx.x;
  const int d  = i % D_INNER;
  const int bl = i / D_INNER;
  float s = dtb[d];
  #pragma unroll
  for (int r = 0; r < DT_RANK; ++r)
    s += dbc[(size_t)bl * 64 + r] * dtw[d * DT_RANK + r];
  dlt[i] = (s > 20.f) ? s : log1pf(__expf(s));
}

// ---------------------------------------------------------------------------
// Selective scan, wave32: 16-lane half-wave per (b,d) channel, lane%16 = state.
// Butterfly reduce over states; fused u*D and SiLU(z) gating.
// ---------------------------------------------------------------------------
__global__ __launch_bounds__(256) void scan_k(
    const float* __restrict__ dlt, const float* __restrict__ uact,
    const float* __restrict__ dbc, const float* __restrict__ A_log,
    const float* __restrict__ Dv,  const float* __restrict__ xz,
    float* __restrict__ y)
{
  const int tid  = threadIdx.x;
  const int lane = tid & 31;
  const int wave = tid >> 5;
  const int n    = lane & 15;
  const int pair = (blockIdx.x * 8 + wave) * 2 + (lane >> 4);
  const int b = pair / D_INNER;
  const int d = pair % D_INNER;
  const float Ac = -__expf(A_log[d * D_STATE + n]);
  const float Dd = Dv[d];
  float h = 0.f;
  const size_t blbase = (size_t)b * SEQLEN;
  for (int l = 0; l < SEQLEN; ++l) {
    const size_t bl = blbase + l;
    const float dv = dlt[bl * D_INNER + d];
    const float uv = uact[bl * D_INNER + d];
    const float Bv = dbc[bl * 64 + DT_RANK + n];
    const float Cv = dbc[bl * 64 + DT_RANK + D_STATE + n];
    h = __expf(dv * Ac) * h + dv * Bv * uv;
    float yp = h * Cv;
    yp += __shfl_xor(yp, 1);
    yp += __shfl_xor(yp, 2);
    yp += __shfl_xor(yp, 4);
    yp += __shfl_xor(yp, 8);
    if (n == 0) {
      const float zv = xz[bl * (2 * D_INNER) + D_INNER + d];
      const float g  = zv / (1.f + __expf(-zv));
      y[bl * D_INNER + d] = (yp + uv * Dd) * g;
    }
  }
}

// ---------------------------------------------------------------------------
// 3x3 conv (64->3) + bias + sigmoid over one batch slice of d (64,512,512).
// Interior tiles: halo staged by the Tensor Data Mover (tensor_load_to_lds,
// 3-D tile 18x18x8, TENSORcnt). Edge tiles: predicated manual path (zero pad).
// ---------------------------------------------------------------------------
__global__ __launch_bounds__(256) void dec_conv_sig_k(
    const float* __restrict__ din, const float* __restrict__ w,
    const float* __restrict__ wb, float* __restrict__ out, int b)
{
  __shared__ float sm[8][18][18];   // TDM tile layout: x fastest, then y, then c
  __shared__ float wsm[3][64][9];
  const int tid = threadIdx.x;
  const int tx = tid & 15, ty = tid >> 4;
  const int x0 = blockIdx.x * 16, y0 = blockIdx.y * 16;
  const bool interior = (blockIdx.x > 0) && (blockIdx.x < 31) &&
                        (blockIdx.y > 0) && (blockIdx.y < 31);
  (void)interior;
  for (int i = tid; i < 3 * 64 * 9; i += 256) {
    const int o = i / 576, rem = i % 576;
    wsm[o][rem / 9][rem % 9] = w[i];
  }
  float s0 = wb[0], s1 = wb[1], s2 = wb[2];
  for (int cc0 = 0; cc0 < 64; cc0 += 8) {
    __syncthreads();
#if HAVE_TDM
    if (interior) {
      if (tid == 0) {
        // D# per CDNA5 ISA: 3-D tensor (64,512,512) f32,
        // tile 18x18x8 starting at (cc0, y0-1, x0-1).
        const unsigned long long ga = (unsigned long long)(uintptr_t)(
            din + ((size_t)cc0 * 512 + (y0 - 1)) * 512 + (x0 - 1));
        const unsigned int lo = (unsigned int)(uintptr_t)(&sm[0][0][0]);
        u32x4 g0; i32x8 g1; i32x4 g2, g3; i32x8 g4;
        g0[0] = 1u;                                   // count=1, user mode
        g0[1] = lo;                                   // lds_addr
        g0[2] = (unsigned int)ga;                     // global_addr[31:0]
        g0[3] = (unsigned int)((ga >> 32) & 0x01FFFFFFu) | (2u << 30); // type=2
        g1[0] = (int)(2u << 16);                      // data_size=4B, mask=0
        g1[1] = (int)(512u << 16);                    // tensor_dim0[15:0]
        g1[2] = (int)(512u << 16);                    // dim0 hi | tensor_dim1 lo
        g1[3] = (int)(18u << 16);                     // dim1 hi | tile_dim0
        g1[4] = (int)(18u | (8u << 16));              // tile_dim1 | tile_dim2
        g1[5] = 512;                                  // tensor_dim0_stride lo32
        g1[6] = 0;                                    // stride0 hi | stride1 lo16
        g1[7] = 4;                                    // tensor_dim1_stride[47:16]
        g2[0] = 64;                                   // tensor_dim2
        g2[1] = 0;                                    // tensor_dim3
        g2[2] = 262144;                               // tensor_dim2_stride lo32
        g2[3] = 0;                                    // stride2 hi | tile_dim3
        g3[0] = 0; g3[1] = 0; g3[2] = 0; g3[3] = 0;
        g4[0] = 0; g4[1] = 0; g4[2] = 0; g4[3] = 0;
        g4[4] = 0; g4[5] = 0; g4[6] = 0; g4[7] = 0;
        __builtin_amdgcn_tensor_load_to_lds(g0, g1, g2, g3, g4, 0);
        __builtin_amdgcn_s_wait_tensorcnt(0);
      }
    } else
#endif
    {
      for (int i = tid; i < 8 * 18 * 18; i += 256) {
        const int cc = i / 324, pos = i % 324;
        const int hy = pos / 18, hx = pos % 18;
        const int gy = y0 - 1 + hy, gx = x0 - 1 + hx;
        float v = 0.f;
        if (gy >= 0 && gy < 512 && gx >= 0 && gx < 512)
          v = din[((size_t)(cc0 + cc) * 512 + gy) * 512 + gx];
        sm[cc][hy][hx] = v;
      }
    }
    __syncthreads();
    #pragma unroll
    for (int cc = 0; cc < 8; ++cc) {
      const int c = cc0 + cc;
      #pragma unroll
      for (int dy = 0; dy < 3; ++dy)
        #pragma unroll
        for (int dx = 0; dx < 3; ++dx) {
          const float v = sm[cc][ty + dy][tx + dx];
          s0 += v * wsm[0][c][dy * 3 + dx];
          s1 += v * wsm[1][c][dy * 3 + dx];
          s2 += v * wsm[2][c][dy * 3 + dx];
        }
    }
  }
  const size_t base = ((size_t)(b * 3) * 512 + (y0 + ty)) * 512 + (x0 + tx);
  out[base]                 = 1.f / (1.f + __expf(-s0));
  out[base + 512 * 512]     = 1.f / (1.f + __expf(-s1));
  out[base + 2 * 512 * 512] = 1.f / (1.f + __expf(-s2));
}

// ---------------------------------------------------------------------------
extern "C" void kernel_launch(void* const* d_in, const int* in_sizes, int n_in,
                              void* d_out, int out_size, void* d_ws, size_t ws_size,
                              hipStream_t stream)
{
  (void)in_sizes; (void)n_in; (void)out_size; (void)ws_size;
  const float* x       = (const float*)d_in[0];
  const float* patch_w = (const float*)d_in[1];   // (192, 768)   N,K
  const float* patch_b = (const float*)d_in[2];
  const float* inprojw = (const float*)d_in[3];   // (768, 192)   N,K
  const float* c1w     = (const float*)d_in[4];
  const float* c1b     = (const float*)d_in[5];
  const float* xprojw  = (const float*)d_in[6];   // (44, 384)    N,K
  const float* dtw     = (const float*)d_in[7];
  const float* dtb     = (const float*)d_in[8];
  const float* A_log   = (const float*)d_in[9];
  const float* Dvec    = (const float*)d_in[10];
  const float* outw    = (const float*)d_in[11];  // (192, 384)   N,K
  const float* dcw     = (const float*)d_in[12];  // (192, 16384) K,N
  const float* dcb     = (const float*)d_in[13];
  const float* fw      = (const float*)d_in[14];
  const float* fb      = (const float*)d_in[15];
  float* out = (float*)d_out;

  float* ws   = (float*)d_ws;
  float* seq  = ws;                  // 8192*192
  float* xz   = seq  + 8192 * 192;   // 8192*768
  float* uact = xz   + 8192 * 768;   // 8192*384
  float* dbc  = uact + 8192 * 384;   // 8192*64 (44 used, ld=64)
  float* dlt  = dbc  + 8192 * 64;    // 8192*384
  float* yb   = dlt  + 8192 * 384;   // 8192*384
  float* sout = yb   + 8192 * 384;   // 8192*192
  float* d_b  = sout + 8192 * 192;   // 64*512*512 (one batch, L2-resident)

  // 1. patch embed as im2col GEMM: M=8192 N=192 K=768 (+bias)
  gemm_wmma<A_IM2COL, EPI_STORE_BIAS, B_NK><<<dim3(3, 64), 256, 0, stream>>>(
      x, patch_w, seq, patch_b, 8192, 192, 768, 192, 768);
  // 2. in_proj: M=8192 N=768 K=192
  gemm_wmma<A_PLAIN, EPI_STORE, B_NK><<<dim3(12, 64), 256, 0, stream>>>(
      seq, inprojw, xz, nullptr, 8192, 768, 192, 768, 192);
  // 3. depthwise causal conv + SiLU
  conv1d_silu_k<<<12288, 256, 0, stream>>>(xz, c1w, c1b, uact);
  // 4. x_proj: M=8192 N=44 K=384 (ld=64 padded)
  gemm_wmma<A_PLAIN, EPI_STORE, B_NK><<<dim3(1, 64), 256, 0, stream>>>(
      uact, xprojw, dbc, nullptr, 8192, 44, 384, 64, 384);
  // 5. dt_proj + softplus
  delta_k<<<12288, 256, 0, stream>>>(dbc, dtw, dtb, dlt);
  // 6. selective scan + gating
  scan_k<<<192, 256, 0, stream>>>(dlt, uact, dbc, A_log, Dvec, xz, yb);
  // 7. out_proj: M=8192 N=192 K=384
  gemm_wmma<A_PLAIN, EPI_STORE, B_NK><<<dim3(3, 64), 256, 0, stream>>>(
      yb, outw, sout, nullptr, 8192, 192, 384, 192, 384);
  // 8. per-batch: deconv GEMM (M=1024 N=16384 K=192, fused bias+ReLU scatter,
  //    64 MB slice stays L2-resident), then TDM-staged 3x3 conv + sigmoid
  for (int b = 0; b < 8; ++b) {
    gemm_wmma<A_PLAIN, EPI_DECONV, B_KN><<<dim3(256, 8), 256, 0, stream>>>(
        sout + (size_t)b * 1024 * 192, dcw, d_b, dcb, 1024, 16384, 192, 0, 16384);
    dec_conv_sig_k<<<dim3(32, 32), 256, 0, stream>>>(d_b, fw, fb, out, b);
  }
}
